// DualPathRNN_17660905521255
// MI455X (gfx1250) — matrix operand order repaired and checked by reading, not compiled, not run on hardware
//
#include <hip/hip_runtime.h>

// ---------------------------------------------------------------------------
// DualPathRNN for MI455X (gfx1250): 12-layer residual LSTM stack
//   C=256, B=16, T=L=4000, gates G=4C=1024, IL=10
// Phase A (parallel): gx = x @ w_ih^T + b_ih + b_hh  -> bf16 WMMA GEMM,
//                     stored transposed as (t, n, b) so the scan loads it
//                     with contiguous b128 accesses.
// Phase B (sequential): persistent 1-workgroup scan, w_hh resident in VGPRs,
//                       h in LDS, v_wmma_f32_16x16x32_bf16 per step.
// ---------------------------------------------------------------------------

typedef __bf16 bf16_t;
typedef __attribute__((ext_vector_type(16))) __bf16 v16bf;
typedef __attribute__((ext_vector_type(8)))  __bf16 v8bf;
typedef __attribute__((ext_vector_type(8)))  float  v8f;
typedef __attribute__((ext_vector_type(4)))  float  v4f;

union V16 { v16bf v; v8bf h[2]; };
union V8F { v8f v; v4f q[2]; };

#define TT   4000
#define BB   16
#define CC   256
#define GG   1024
#define MM   (TT * BB)          // 64000 rows
#define ROW  (BB * CC)          // 4096 floats per time step (x/y)
#define GROW (BB * GG)          // 16384 floats per time step (gates)

__device__ __forceinline__ float fsig(float x) {
    return 1.0f / (1.0f + __expf(-x));
}
__device__ __forceinline__ float ftanh(float x) {
    return 2.0f / (1.0f + __expf(-2.0f * x)) - 1.0f;
}

// -------------------- layout / conversion kernels --------------------------

// x (B, C, L) fp32 -> xs (T, B, C) fp32 and xb (T, B, C) bf16
__global__ void k_transpose_in(const float* __restrict__ x,
                               float* __restrict__ xs, bf16_t* __restrict__ xb) {
    size_t e = (size_t)blockIdx.x * blockDim.x + threadIdx.x;
    if (e >= (size_t)MM * CC) return;
    int t = (int)(e >> 12);
    int r = (int)(e & 4095);
    int b = r >> 8, c = r & 255;
    float v = x[((size_t)b * CC + c) * TT + t];
    xs[e] = v;
    xb[e] = (bf16_t)v;
}

// fp32 weights -> bf16 (both w_ih and w_hh for current layer)
__global__ void k_cvt_w(const float* __restrict__ wih, const float* __restrict__ whh,
                        bf16_t* __restrict__ wihb, bf16_t* __restrict__ whhb) {
    int i = blockIdx.x * blockDim.x + threadIdx.x;
    if (i >= GG * CC) return;
    wihb[i] = (bf16_t)wih[i];
    whhb[i] = (bf16_t)whh[i];
}

// residual + optional time reversal; regenerates bf16 copy for next GEMM
__global__ void k_residual(const float* __restrict__ xin, const float* __restrict__ y,
                           float* __restrict__ xout, bf16_t* __restrict__ xb, int rev) {
    size_t e = (size_t)blockIdx.x * blockDim.x + threadIdx.x;
    if (e >= (size_t)MM * CC) return;
    int t = (int)(e >> 12);
    int r = (int)(e & 4095);
    int src = rev ? (TT - 1 - t) : t;
    size_t si = (size_t)src * ROW + r;
    float v = xin[si] + y[si];
    xout[e] = v;
    xb[e] = (bf16_t)v;
}

// xs (T, B, C) -> out (B, C, L)
__global__ void k_transpose_out(const float* __restrict__ xs, float* __restrict__ out) {
    size_t e = (size_t)blockIdx.x * blockDim.x + threadIdx.x;
    if (e >= (size_t)MM * CC) return;
    int t = (int)(e >> 12);
    int r = (int)(e & 4095);
    int b = r >> 8, c = r & 255;
    out[((size_t)b * CC + c) * TT + t] = xs[e];
}

// -------------------- input-projection GEMM (WMMA bf16) ---------------------
// gx_t[t, n, b] = sum_k xb[t*16+b, k] * wih[n, k] + b_ih[n] + b_hh[n]
// One wave computes a 16(M=b within t) x 64(N) tile; 8 waves per block.
// Output stored (t, n, b): per lane, b = vgpr + 8*hi is CONTIGUOUS -> b128.
__global__ __launch_bounds__(256)
void k_gemm_gx(const bf16_t* __restrict__ xb, const bf16_t* __restrict__ wih,
               const float* __restrict__ b_ih, const float* __restrict__ b_hh,
               float* __restrict__ gx) {
    int gwave = (blockIdx.x * 256 + threadIdx.x) >> 5;
    int lane  = threadIdx.x & 31;
    int mtile = gwave >> 4;      // time step t, 0..3999
    int nsup  = gwave & 15;      // 0..15 (64 columns each)
    int nrow  = lane & 15;
    int hi    = lane >> 4;

    v8f acc[4];
#pragma unroll
    for (int j = 0; j < 4; ++j)
#pragma unroll
        for (int r = 0; r < 8; ++r) acc[j][r] = 0.0f;

    const bf16_t* arow = xb + (size_t)(mtile * 16 + nrow) * CC;
#pragma unroll
    for (int k = 0; k < 8; ++k) {
        V16 a;   // A 16x32 bf16 fragment: K = 32k + hi*16 + [0,16)
        a.h[0] = *(const v8bf*)(arow + 32 * k + hi * 16);
        a.h[1] = *(const v8bf*)(arow + 32 * k + hi * 16 + 8);
#pragma unroll
        for (int j = 0; j < 4; ++j) {
            int ncol = nsup * 64 + j * 16 + nrow;
            const bf16_t* wrow = wih + (size_t)ncol * CC + 32 * k + hi * 16;
            V16 b;   // B 32x16 fragment: lane = N, 16 contiguous K per half-wave
            b.h[0] = *(const v8bf*)(wrow);
            b.h[1] = *(const v8bf*)(wrow + 8);
            acc[j] = __builtin_amdgcn_wmma_f32_16x16x32_bf16(
                false, a.v, false, b.v, (short)0, acc[j], false, false);
        }
    }
#pragma unroll
    for (int j = 0; j < 4; ++j) {
        int ncol = nsup * 64 + j * 16 + nrow;
        float bias = b_ih[ncol] + b_hh[ncol];
        float* gout = gx + ((size_t)mtile * GG + ncol) * 16 + hi * 8;
        v4f lo = { acc[j][0] + bias, acc[j][1] + bias,
                   acc[j][2] + bias, acc[j][3] + bias };
        v4f hq = { acc[j][4] + bias, acc[j][5] + bias,
                   acc[j][6] + bias, acc[j][7] + bias };
        *(v4f*)(gout)     = lo;
        *(v4f*)(gout + 4) = hq;
    }
}

// -------------------- persistent recurrent scan -----------------------------
// 1 workgroup, 512 threads = 16 waves. Wave w owns gate columns [64w, 64w+64).
// w_hh fragments live in VGPRs for the entire 4000-step loop.
__global__ __launch_bounds__(512, 1)
void k_scan(const float* __restrict__ gx, const bf16_t* __restrict__ whh,
            float* __restrict__ y, int permute_out) {
    __shared__ float  lds_gates[BB * GG];   // 64 KB
    __shared__ bf16_t lds_h[BB * CC];       // 8 KB   (total 72 KB < 320 KB/WGP)

    const int tid  = threadIdx.x;
    const int wave = tid >> 5;
    const int lane = tid & 31;
    const int nrow = lane & 15;
    const int hi   = lane >> 4;

    // Preload this wave's w_hh B-fragments into registers: 4 tiles x 8 K-steps
    V16 wf[4][8];
#pragma unroll
    for (int j = 0; j < 4; ++j) {
        int ncol = wave * 64 + j * 16 + nrow;
        const bf16_t* wrow = whh + (size_t)ncol * CC;
#pragma unroll
        for (int k = 0; k < 8; ++k) {
            wf[j][k].h[0] = *(const v8bf*)(wrow + 32 * k + hi * 16);
            wf[j][k].h[1] = *(const v8bf*)(wrow + 32 * k + hi * 16 + 8);
        }
    }

    // h = 0, c = 0
#pragma unroll
    for (int j = 0; j < 8; ++j) lds_h[tid + 512 * j] = (bf16_t)0.0f;
    float cst[8];
#pragma unroll
    for (int j = 0; j < 8; ++j) cst[j] = 0.0f;
    __syncthreads();

    for (int s = 0; s < TT; ++s) {
        if (s + 1 < TT)   // pull next step's gate rows toward L2/WGP$
            __builtin_prefetch(gx + (size_t)(s + 1) * GROW + tid * 32, 0, 3);

        // acc <- gx_t[s, n, b]: per lane 8 contiguous floats -> 2x b128 loads
        V8F acc[4];
        const float* grow = gx + (size_t)s * GROW + hi * 8;
#pragma unroll
        for (int j = 0; j < 4; ++j) {
            int ncol = wave * 64 + j * 16 + nrow;
            const float* gp = grow + (size_t)ncol * 16;
            acc[j].q[0] = *(const v4f*)(gp);
            acc[j].q[1] = *(const v4f*)(gp + 4);
        }

        // acc += h @ w_hh^T  (A fragment loaded once per K-step, reused 4x)
#pragma unroll
        for (int k = 0; k < 8; ++k) {
            V16 a;
            const bf16_t* hrow = lds_h + nrow * CC + 32 * k + hi * 16;
            a.h[0] = *(const v8bf*)(hrow);
            a.h[1] = *(const v8bf*)(hrow + 8);
#pragma unroll
            for (int j = 0; j < 4; ++j)
                acc[j].v = __builtin_amdgcn_wmma_f32_16x16x32_bf16(
                    false, a.v, false, wf[j][k].v, (short)0, acc[j].v, false, false);
        }

        // gates -> LDS (row b = r + 8*hi, col n = ncol)
#pragma unroll
        for (int j = 0; j < 4; ++j) {
            int ncol = wave * 64 + j * 16 + nrow;
#pragma unroll
            for (int r = 0; r < 8; ++r)
                lds_gates[(r + hi * 8) * GG + ncol] = acc[j].v[r];
        }
        __syncthreads();

        // activations: each thread owns 8 fixed (b,c) cells (idx = tid + 512j:
        // stride-1 LDS banks, coalesced y stores), c-state carried in regs
        int dst = permute_out ? ((s % 400) * 10 + s / 400) : s;
        float* yrow = y + (size_t)dst * ROW;
#pragma unroll
        for (int j = 0; j < 8; ++j) {
            int idx = tid + 512 * j;
            int b = idx >> 8, c = idx & 255;
            const float* gb = lds_gates + b * GG;
            float ig = fsig(gb[c]);
            float fg = fsig(gb[256 + c]);
            float gg = ftanh(gb[512 + c]);
            float og = fsig(gb[768 + c]);
            float cs = fg * cst[j] + ig * gg;
            cst[j] = cs;
            float h = og * ftanh(cs);
            yrow[idx] = h;
            lds_h[idx] = (bf16_t)h;
        }
        __syncthreads();
    }
}

// ---------------------------------------------------------------------------

extern "C" void kernel_launch(void* const* d_in, const int* in_sizes, int n_in,
                              void* d_out, int out_size, void* d_ws, size_t ws_size,
                              hipStream_t stream) {
    const float* x    = (const float*)d_in[0];
    const float* w_ih = (const float*)d_in[1];
    const float* w_hh = (const float*)d_in[2];
    const float* b_ih = (const float*)d_in[3];
    const float* b_hh = (const float*)d_in[4];
    float* out = (float*)d_out;

    char* ws = (char*)d_ws;
    size_t off = 0;
    auto alloc = [&](size_t bytes) -> void* {
        void* p = ws + off;
        off += (bytes + 255) & ~(size_t)255;
        return p;
    };
    float*  xs0  = (float*) alloc((size_t)MM * CC * 4);
    float*  xs1  = (float*) alloc((size_t)MM * CC * 4);
    bf16_t* xb   = (bf16_t*)alloc((size_t)MM * CC * 2);
    float*  gx   = (float*) alloc((size_t)MM * GG * 4);
    float*  ybuf = (float*) alloc((size_t)MM * CC * 4);
    bf16_t* wihb = (bf16_t*)alloc((size_t)GG * CC * 2);
    bf16_t* whhb = (bf16_t*)alloc((size_t)GG * CC * 2);
    (void)ws_size; (void)in_sizes; (void)n_in; (void)out_size;

    const size_t NE = (size_t)MM * CC;          // 16,384,000 elements
    const int eb = 256;
    const int egrid = (int)((NE + eb - 1) / eb);

    k_transpose_in<<<egrid, eb, 0, stream>>>(x, xs0, xb);

    float* xcur = xs0;
    float* xnxt = xs1;
    for (int l = 0; l < 12; ++l) {
        k_cvt_w<<<(GG * CC) / 256, 256, 0, stream>>>(
            w_ih + (size_t)l * GG * CC, w_hh + (size_t)l * GG * CC, wihb, whhb);

        // 4000 M-tiles x 16 N-supertiles = 64000 waves; 8 waves/block
        k_gemm_gx<<<8000, 256, 0, stream>>>(xb, wihb, b_ih + (size_t)l * GG,
                                            b_hh + (size_t)l * GG, gx);

        k_scan<<<1, 512, 0, stream>>>(gx, whhb, ybuf, (l % 2 == 0) ? 1 : 0);

        k_residual<<<egrid, eb, 0, stream>>>(xcur, ybuf, xnxt, xb,
                                             (l % 2 == 1) ? 1 : 0);
        float* tmp = xcur; xcur = xnxt; xnxt = tmp;
    }

    k_transpose_out<<<egrid, eb, 0, stream>>>(xcur, out);
}
